// Precoding_gnn_44495861186889
// MI455X (gfx1250) — matrix operand order repaired and checked by reading, not compiled, not run on hardware
//
#include <hip/hip_runtime.h>

typedef __attribute__((ext_vector_type(16))) _Float16 v16h;
typedef __attribute__((ext_vector_type(8)))  _Float16 v8h;
typedef __attribute__((ext_vector_type(2)))  _Float16 f16x2;
typedef __attribute__((ext_vector_type(8)))  float    v8f;

#define TPB 256
#define MM  64
#define KK  32
#define DD  32

// f16 weight pool offsets (elements)
#define OFF_W1S 0
#define OFF_W1M 64
#define OFF_W1K 128
#define OFF_W2S 192
#define OFF_W2M (192 + 1024)
#define OFF_W2K (192 + 2048)
#define OFF_W3S (192 + 3072)
#define OFF_W3M (192 + 4096)
#define OFF_W3K (192 + 5120)
#define OFF_W4S (192 + 6144)
#define OFF_W4M (192 + 7168)
#define OFF_W4K (192 + 8192)
#define OFF_W5S (192 + 9216)
#define OFF_W5M (192 + 9216 + 64)
#define OFF_W5K (192 + 9216 + 128)
#define WPOOL   (192 + 9216 + 192)   // 9600 f16 elements

struct Smem {
  _Float16 H0[MM * KK * DD];     // 128 KB (head aliased as f32 Out, 16 KB)
  _Float16 H1[MM * KK * DD];     // 128 KB (head aliased as f32 Xs, 16 KB)
  _Float16 msgMh[MM * DD];       // 4 KB   f16 aggregates (WMMA B operands)
  _Float16 msgKh[KK * DD];       // 2 KB
  float    BM[MM * DD];          // 8 KB   broadcast bias  BM[m][h]
  float    BK[KK * DD];          // 4 KB   broadcast bias  BK[k][h]
  _Float16 Wf[WPOOL];            // 18.75 KB f16 weights
  float    red[TPB];             // 1 KB
};                               // ~294 KB  (<= 320 KB/WGP)

__device__ __forceinline__ v16h cat16(v8h lo, v8h hi) {
  return __builtin_shufflevector(lo, hi, 0,1,2,3,4,5,6,7,8,9,10,11,12,13,14,15);
}

// A fragment: 16 weight rows starting at hbase; ISA A-layout (K interleaved by lane half).
__device__ __forceinline__ v16h load_afrag(const _Float16* W, int hbase, int l15, int half) {
  const _Float16* row = W + (hbase + l15) * DD;
  const int ko = half * 8;
  v8h lo = *(const v8h*)(row + ko);        // VGPR0-3: d = ko..ko+7
  v8h hi = *(const v8h*)(row + 16 + ko);   // VGPR4-7: d = 16+ko..16+ko+7
  return cat16(lo, hi);
}

// B fragment: lane's column entity; contraction d interleaved by lane half (same K order as A).
__device__ __forceinline__ v16h load_bfrag(const _Float16* entityRow, int half) {
  const int kb2 = half * 8;
  v8h lo = *(const v8h*)(entityRow + kb2);       // d = kb2..kb2+7
  v8h hi = *(const v8h*)(entityRow + kb2 + 16);  // d = 16+kb2..16+kb2+7
  return cat16(lo, hi);
}

// Middle GNN layer (32 -> 32): f16-pk aggregates -> WMMA biases -> WMMA GEMM (+bias, ReLU).
// All WMMAs computed transposed: D[h][entity], so every LDS access is contiguous b128.
__device__ __forceinline__ void midlayer(Smem& s,
                                         const _Float16* __restrict__ Hin,
                                         _Float16* __restrict__ Hout,
                                         int wsOff, int wmOff, int wkOff, int tid)
{
  const int lane = tid & 31, wv = tid >> 5, half = lane >> 4, l15 = lane & 15;

  // ---- Step A: aggregates (packed f16 adds, two interleaved accumulators) ----
  for (int pe = tid; pe < (MM * DD) / 2; pe += TPB) {          // msg_m: 1024 pair-cols
    int m = pe >> 4, dp = (pe & 15) * 2;
    const f16x2* col = (const f16x2*)(Hin + m * KK * DD + dp); // k stride = 16 f16x2
    f16x2 a0 = col[0], a1 = col[16];
    #pragma unroll
    for (int k = 2; k < KK; k += 2) { a0 += col[k * 16]; a1 += col[k * 16 + 16]; }
    *(f16x2*)(s.msgMh + m * DD + dp) = a0 + a1;
  }
  for (int pe = tid; pe < (KK * DD) / 2; pe += TPB) {          // msg_k: 512 pair-cols
    int k = pe >> 4, dp = (pe & 15) * 2;
    const f16x2* col = (const f16x2*)(Hin + k * DD + dp);      // m stride = 512 f16x2
    f16x2 a0 = col[0], a1 = col[512];
    #pragma unroll
    for (int m = 2; m < MM; m += 2) { a0 += col[m * 512]; a1 += col[m * 512 + 512]; }
    *(f16x2*)(s.msgKh + k * DD + dp) = a0 + a1;
  }
  __syncthreads();

  // ---- Step B: biases via WMMA.  BM: 8 tiles (all waves), BK: 4 tiles (waves 0-3) ----
  {
    const v8f zero = {};
    const int ht = wv >> 2, mB = wv & 3;                 // D[h-tile ht][m-tile mB]
    v16h aW = load_afrag(s.Wf + wmOff, ht * 16, l15, half);
    v16h bM = load_bfrag(s.msgMh + (mB * 16 + l15) * DD, half);
    v8f c = __builtin_amdgcn_wmma_f32_16x16x32_f16(false, aW, false, bM,
                                                   (short)0, zero, false, false);
    float* dst = s.BM + (mB * 16 + l15) * DD + ht * 16 + half * 8;
    #pragma unroll
    for (int j = 0; j < 8; ++j) dst[j] = c[j];
    if (wv < 4) {                                        // wave-uniform branch
      const int ht2 = wv >> 1, kB = wv & 1;
      v16h aK = load_afrag(s.Wf + wkOff, ht2 * 16, l15, half);
      v16h bK = load_bfrag(s.msgKh + (kB * 16 + l15) * DD, half);
      v8f c2 = __builtin_amdgcn_wmma_f32_16x16x32_f16(false, aK, false, bK,
                                                      (short)0, zero, false, false);
      float* dst2 = s.BK + (kB * 16 + l15) * DD + ht2 * 16 + half * 8;
      #pragma unroll
      for (int j = 0; j < 8; ++j) dst2[j] = c2[j];
    }
  }
  __syncthreads();

  // ---- Step C: z = Ws·Hᵀ + BM + BK, ReLU.  D[h][p]; 16 tiles per wave ----
  const v16h aS0 = load_afrag(s.Wf + wsOff, 0,  l15, half);   // h-tile 0, reused 16x
  const v16h aS1 = load_afrag(s.Wf + wsOff, 16, l15, half);   // h-tile 1
  for (int i = wv * 16; i < wv * 16 + 16; ++i) {
    const int m = i >> 1, kb = (i & 1) * 16;
    const int p = m * KK + kb + l15;                     // lane's position (column)
    const v16h bH = load_bfrag(Hin + p * DD, half);
    #pragma unroll
    for (int ht = 0; ht < 2; ++ht) {
      const int hb = ht * 16 + half * 8;
      const float* bmrow = s.BM + m * DD + hb;           // 8 contiguous f32
      const float* bkrow = s.BK + (kb + l15) * DD + hb;  // 8 contiguous f32
      v8f c;
      #pragma unroll
      for (int j = 0; j < 8; ++j) c[j] = bmrow[j] + bkrow[j];
      c = __builtin_amdgcn_wmma_f32_16x16x32_f16(false, (ht ? aS1 : aS0), false, bH,
                                                 (short)0, c, false, false);
      v8h hv;
      #pragma unroll
      for (int j = 0; j < 8; ++j) {
        float z = c[j] > 0.f ? c[j] : 0.f;
        hv[j] = (_Float16)z;
      }
      *(v8h*)(Hout + p * DD + hb) = hv;                  // one ds_store_b128
    }
  }
  __syncthreads();
}

__global__ __launch_bounds__(TPB, 1)
void Precoding_gnn_44495861186889_kernel(
    const float* __restrict__ x,
    const float* w1s, const float* w1m, const float* w1k,
    const float* w2s, const float* w2m, const float* w2k,
    const float* w3s, const float* w3m, const float* w3k,
    const float* w4s, const float* w4m, const float* w4k,
    const float* w5s, const float* w5m, const float* w5k,
    float* __restrict__ out)
{
  __shared__ Smem s;
  const int tid = threadIdx.x;
  const int b   = blockIdx.x;

  // Stage x (f32) into H1 head (free until layer 2 writes H1).
  float* Xs = reinterpret_cast<float*>(s.H1);
  const float* xb = x + (size_t)b * (MM * KK * 2);
  for (int i = tid; i < MM * KK * 2; i += TPB) Xs[i] = xb[i];

  // Weights -> f16 LDS pool.
  auto cvt = [&](const float* src, int off, int n) {
    for (int i = tid; i < n; i += TPB) s.Wf[off + i] = (_Float16)src[i];
  };
  cvt(w1s, OFF_W1S, 64);   cvt(w1m, OFF_W1M, 64);   cvt(w1k, OFF_W1K, 64);
  cvt(w2s, OFF_W2S, 1024); cvt(w2m, OFF_W2M, 1024); cvt(w2k, OFF_W2K, 1024);
  cvt(w3s, OFF_W3S, 1024); cvt(w3m, OFF_W3M, 1024); cvt(w3k, OFF_W3K, 1024);
  cvt(w4s, OFF_W4S, 1024); cvt(w4m, OFF_W4M, 1024); cvt(w4k, OFF_W4K, 1024);
  cvt(w5s, OFF_W5S, 64);   cvt(w5m, OFF_W5M, 64);   cvt(w5k, OFF_W5K, 64);
  __syncthreads();

  // ---------------- layer 1 (d_in = 2): scalar, packed f16 output ----------------
  // f32 aggregates live in BM/BK scratch.
  for (int idx = tid; idx < MM * 2; idx += TPB) {
    int m = idx >> 1, c = idx & 1;
    float acc = 0.f;
    #pragma unroll
    for (int k = 0; k < KK; ++k) acc += Xs[(m * KK + k) * 2 + c];
    s.BM[idx] = acc;
  }
  for (int idx = tid; idx < KK * 2; idx += TPB) {
    int k = idx >> 1, c = idx & 1;
    float acc = 0.f;
    #pragma unroll
    for (int m = 0; m < MM; ++m) acc += Xs[(m * KK + k) * 2 + c];
    s.BK[idx] = acc;
  }
  __syncthreads();
  for (int p = tid; p < MM * KK; p += TPB) {
    int m = p >> 5, k = p & 31;
    float x0 = Xs[p * 2], x1 = Xs[p * 2 + 1];
    float a0 = s.BM[m * 2], a1 = s.BM[m * 2 + 1];
    float g0 = s.BK[k * 2], g1 = s.BK[k * 2 + 1];
    #pragma unroll
    for (int hq = 0; hq < 4; ++hq) {
      v8h hv;
      #pragma unroll
      for (int j = 0; j < 8; ++j) {
        int h = hq * 8 + j;
        float z = (float)s.Wf[OFF_W1S + h * 2] * x0 + (float)s.Wf[OFF_W1S + h * 2 + 1] * x1
                + (float)s.Wf[OFF_W1M + h * 2] * a0 + (float)s.Wf[OFF_W1M + h * 2 + 1] * a1
                + (float)s.Wf[OFF_W1K + h * 2] * g0 + (float)s.Wf[OFF_W1K + h * 2 + 1] * g1;
        z = z > 0.f ? z : 0.f;
        hv[j] = (_Float16)z;
      }
      *(v8h*)(s.H0 + p * DD + hq * 8) = hv;
    }
  }
  __syncthreads();

  // ---------------- layers 2-4: WMMA ----------------
  midlayer(s, s.H0, s.H1, OFF_W2S, OFF_W2M, OFF_W2K, tid);   // overwrites Xs (ok)
  midlayer(s, s.H1, s.H0, OFF_W3S, OFF_W3M, OFF_W3K, tid);
  midlayer(s, s.H0, s.H1, OFF_W4S, OFF_W4M, OFF_W4K, tid);

  // ---------------- layer 5 (d_out = 2): scalar + power norm ----------------
  const _Float16* Hin = s.H1;
  for (int pe = tid; pe < (MM * DD) / 2; pe += TPB) {
    int m = pe >> 4, dp = (pe & 15) * 2;
    const f16x2* col = (const f16x2*)(Hin + m * KK * DD + dp);
    f16x2 a0 = col[0], a1 = col[16];
    #pragma unroll
    for (int k = 2; k < KK; k += 2) { a0 += col[k * 16]; a1 += col[k * 16 + 16]; }
    *(f16x2*)(s.msgMh + m * DD + dp) = a0 + a1;
  }
  for (int pe = tid; pe < (KK * DD) / 2; pe += TPB) {
    int k = pe >> 4, dp = (pe & 15) * 2;
    const f16x2* col = (const f16x2*)(Hin + k * DD + dp);
    f16x2 a0 = col[0], a1 = col[512];
    #pragma unroll
    for (int m = 2; m < MM; m += 2) { a0 += col[m * 512]; a1 += col[m * 512 + 512]; }
    *(f16x2*)(s.msgKh + k * DD + dp) = a0 + a1;
  }
  __syncthreads();
  for (int idx = tid; idx < MM * 2; idx += TPB) {       // b5M[m][c]
    int m = idx >> 1, c = idx & 1;
    float acc = 0.f;
    #pragma unroll
    for (int d = 0; d < DD; ++d)
      acc += (float)s.Wf[OFF_W5M + c * DD + d] * (float)s.msgMh[m * DD + d];
    s.BM[idx] = acc;
  }
  for (int idx = tid; idx < KK * 2; idx += TPB) {       // b5K[k][c]
    int k = idx >> 1, c = idx & 1;
    float acc = 0.f;
    #pragma unroll
    for (int d = 0; d < DD; ++d)
      acc += (float)s.Wf[OFF_W5K + c * DD + d] * (float)s.msgKh[k * DD + d];
    s.BK[idx] = acc;
  }
  __syncthreads();

  float* Out = reinterpret_cast<float*>(s.H0);          // 16 KB, H0 free now
  float psum = 0.f;
  for (int p = tid; p < MM * KK; p += TPB) {
    int m = p >> 5, k = p & 31;
    float o0 = s.BM[m * 2]     + s.BK[k * 2];
    float o1 = s.BM[m * 2 + 1] + s.BK[k * 2 + 1];
    #pragma unroll
    for (int d = 0; d < DD; ++d) {
      float hv = (float)Hin[p * DD + d];
      o0 += (float)s.Wf[OFF_W5S + d]      * hv;
      o1 += (float)s.Wf[OFF_W5S + DD + d] * hv;
    }
    Out[p * 2]     = o0;
    Out[p * 2 + 1] = o1;
    psum += o0 * o0 + o1 * o1;
  }
  s.red[tid] = psum;
  __syncthreads();
  for (int off = TPB / 2; off > 0; off >>= 1) {
    if (tid < off) s.red[tid] += s.red[tid + off];
    __syncthreads();
  }
  const float alpha = rsqrtf(s.red[0]);                 // PT = 1.0
  float* og = out + (size_t)b * (MM * KK * 2);
  for (int i = tid; i < MM * KK * 2; i += TPB) og[i] = alpha * Out[i];
}

extern "C" void kernel_launch(void* const* d_in, const int* in_sizes, int n_in,
                              void* d_out, int out_size, void* d_ws, size_t ws_size,
                              hipStream_t stream) {
  const float* x = (const float*)d_in[0];
  const int bs = in_sizes[0] / (MM * KK * 2);   // 1024
  Precoding_gnn_44495861186889_kernel<<<dim3(bs), dim3(TPB), 0, stream>>>(
      x,
      (const float*)d_in[1],  (const float*)d_in[2],  (const float*)d_in[3],
      (const float*)d_in[4],  (const float*)d_in[5],  (const float*)d_in[6],
      (const float*)d_in[7],  (const float*)d_in[8],  (const float*)d_in[9],
      (const float*)d_in[10], (const float*)d_in[11], (const float*)d_in[12],
      (const float*)d_in[13], (const float*)d_in[14], (const float*)d_in[15],
      (float*)d_out);
}
